// MultiHeadAttention_26018911879518
// MI455X (gfx1250) — matrix operand order repaired and checked by reading, not compiled, not run on hardware
//
#include <hip/hip_runtime.h>
#include <hip/hip_bf16.h>
#include <stdint.h>

#define DEVFN __device__ __forceinline__

typedef __attribute__((ext_vector_type(4)))  __bf16 bf16x4;
typedef __attribute__((ext_vector_type(8)))  __bf16 bf16x8;
typedef __attribute__((ext_vector_type(16))) __bf16 bf16x16;
typedef __attribute__((ext_vector_type(8)))  float  f32x8;

constexpr int BB   = 4;
constexpr int NSEQ = 2048;
constexpr int CDIM = 1024;
constexpr int HH   = 16;
constexpr int DH   = 64;
constexpr float SM_SCALE = 0.125f;  // 1/sqrt(64)

DEVFN f32x8 zero8() {
  f32x8 z;
#pragma unroll
  for (int i = 0; i < 8; ++i) z[i] = 0.0f;
  return z;
}

// Build a 16x32 A / 32x16 B bf16 fragment from two contiguous 16B LDS reads.
DEVFN bf16x16 ld_frag16(const __bf16* lo, const __bf16* hi) {
  bf16x8 l = *(const bf16x8*)lo;
  bf16x8 h = *(const bf16x8*)hi;
  bf16x16 r;
#pragma unroll
  for (int i = 0; i < 8; ++i) { r[i] = l[i]; r[8 + i] = h[i]; }
  return r;
}

DEVFN f32x8 wmma_bf16(bf16x16 a, bf16x16 b, f32x8 c) {
  return __builtin_amdgcn_wmma_f32_16x16x32_bf16(
      /*neg_a=*/false, a, /*neg_b=*/false, b,
      /*c_mod=*/(short)0, c, /*reuse_a=*/false, /*reuse_b=*/false);
}

// CDNA5 async DMA: copy 16 bytes/lane global -> LDS, tracked by ASYNCcnt.
// Generic shared pointers carry the DS byte address in their low 32 bits.
DEVFN void async_ld_b128(const void* lds_dst, const void* gsrc) {
  asm volatile("global_load_async_to_lds_b128 %0, %1, off"
               :
               : "v"((uint32_t)(uintptr_t)lds_dst), "v"(gsrc)
               : "memory");
}
DEVFN void wait_async0() { asm volatile("s_wait_asynccnt 0x0" ::: "memory"); }

// ---------------------------------------------------------------------------
// One-time fp32 -> bf16 conversion (pays VALU once so all GEMM/attention tile
// stages become pure byte copies, enabling the async-LDS DMA path).
// ---------------------------------------------------------------------------
__global__ __launch_bounds__(256) void cvt_f32_to_bf16(
    const float* __restrict__ src, __bf16* __restrict__ dst, int n4) {
  const int i = blockIdx.x * blockDim.x + threadIdx.x;
  if (i < n4) {
    const float4 v = ((const float4*)src)[i];
    bf16x4 o;
    o[0] = (__bf16)v.x; o[1] = (__bf16)v.y;
    o[2] = (__bf16)v.z; o[3] = (__bf16)v.w;
    ((bf16x4*)dst)[i] = o;
  }
}

// ---------------------------------------------------------------------------
// GEMM: Y[M,Nc] = A[M,K] * W[K,Nc] + bias, all-bf16 inputs, fp32 accumulate.
// Tile: 128(M) x 128(N) per workgroup, 8 waves in 4x2, 32x64 per wave,
// BK=32 == one v_wmma_f32_16x16x32_bf16 per K-step; 8 WMMA / wave / K-step.
// Double-buffered LDS: async DMA of tile i+1 overlaps WMMA of tile i; one
// s_wait_asynccnt + one barrier per K-step.
// OUTMODE 0: bf16 output remapped to [B,H,N,D] (QKV). OUTMODE 1: fp32.
// ---------------------------------------------------------------------------
template <int OUTMODE>
__global__ __launch_bounds__(256) void gemm_bf16_wmma(
    const __bf16* __restrict__ A, const __bf16* __restrict__ W,
    const float* __restrict__ bias, void* __restrict__ outp,
    int M, int K, int Nc) {
  constexpr int BM = 128, BN = 128, BK = 32, LDT = BK + 8;  // 80B row stride
  __shared__ __bf16 As[2][BM][LDT];
  __shared__ __bf16 Bs[2][BN][LDT];  // [n][k] so B-frag reads are contiguous

  const int tid  = threadIdx.x;
  const int lane = tid & 31;
  const int wave = tid >> 5;
  const int nl   = lane & 15;
  const int half = lane >> 4;

  const int m_blk = blockIdx.x * BM;
  const int n_blk = blockIdx.y * BN;
  const int wm = (wave & 3) * 32;   // 4 waves across M
  const int wn = (wave >> 2) * 64;  // 2 waves across N

  auto stage = [&](int buf, int kb) {
    // A tile (128x32 bf16) via async DMA: 16B per lane, 64 rows per issue.
    {
      const int r = tid >> 2, seg = (tid & 3) * 8;
#pragma unroll
      for (int p = 0; p < 2; ++p) {
        const int row = p * 64 + r;
        async_ld_b128(&As[buf][row][seg],
                      &A[(size_t)(m_blk + row) * K + kb + seg]);
      }
    }
    // W tile (32x128) transposed into Bs[n][k].
    {
      const int k  = tid >> 4;        // 0..15
      const int ng = (tid & 15) * 8;  // 0..120
#pragma unroll
      for (int p = 0; p < 2; ++p) {
        const int kk = p * 16 + k;
        const bf16x8 v = *(const bf16x8*)&W[(size_t)(kb + kk) * Nc + n_blk + ng];
#pragma unroll
        for (int j = 0; j < 8; ++j) Bs[buf][ng + j][kk] = v[j];
      }
    }
  };

  f32x8 acc[2][4];
#pragma unroll
  for (int i = 0; i < 2; ++i)
#pragma unroll
    for (int j = 0; j < 4; ++j) acc[i][j] = zero8();

  stage(0, 0);
  int cur = 0;
  for (int kb = 0; kb < K; kb += BK) {
    wait_async0();     // own-wave async copies complete...
    __syncthreads();   // ...then all waves' copies + ds writes visible
    if (kb + BK < K) stage(cur ^ 1, kb + BK);  // overlap next tile DMA

    bf16x16 afr[2], bfr[4];
#pragma unroll
    for (int tm = 0; tm < 2; ++tm) {
      const int row = wm + tm * 16 + nl;
      afr[tm] = ld_frag16(&As[cur][row][half * 8], &As[cur][row][16 + half * 8]);
    }
#pragma unroll
    for (int tn = 0; tn < 4; ++tn) {
      const int col = wn + tn * 16 + nl;
      bfr[tn] = ld_frag16(&Bs[cur][col][half * 8], &Bs[cur][col][16 + half * 8]);
    }
#pragma unroll
    for (int tm = 0; tm < 2; ++tm)
#pragma unroll
      for (int tn = 0; tn < 4; ++tn)
        acc[tm][tn] = wmma_bf16(afr[tm], bfr[tn], acc[tm][tn]);
    cur ^= 1;
  }

  // ---- epilogue: bias + store ----
#pragma unroll
  for (int tm = 0; tm < 2; ++tm) {
#pragma unroll
    for (int tn = 0; tn < 4; ++tn) {
      const int gn = n_blk + wn + tn * 16 + nl;
      const float bv = bias[gn];
#pragma unroll
      for (int r = 0; r < 8; ++r) {
        const int gm = m_blk + wm + tm * 16 + r + half * 8;
        const float val = acc[tm][tn][r] + bv;
        if constexpr (OUTMODE == 0) {
          // remap (token-row, channel) -> [B,H,N,D] bf16
          const int b = gm >> 11, nt = gm & (NSEQ - 1);
          const int h = gn >> 6,  d  = gn & (DH - 1);
          ((__bf16*)outp)[(((size_t)(b * HH + h) * NSEQ + nt) << 6) + d] = (__bf16)val;
        } else {
          ((float*)outp)[(size_t)gm * Nc + gn] = val;
        }
      }
    }
  }
}

// ---------------------------------------------------------------------------
// Flash attention forward. One workgroup = 128 queries (8 waves x 16 rows)
// for one (b,h). K/V blocks double-buffered in LDS (K via async DMA, V staged
// transposed); one barrier per key block. Online softmax state in VGPRs; the
// P tile round-trips through per-wave LDS (wave-local s_wait_dscnt fence) to
// convert the C-layout score fragment into an A-layout fragment for P@V.
// ---------------------------------------------------------------------------
__global__ __launch_bounds__(256) void attn_fwd(
    const __bf16* __restrict__ Qg, const __bf16* __restrict__ Kg,
    const __bf16* __restrict__ Vg, __bf16* __restrict__ Og) {
  constexpr int LQ = DH + 8;  // 144B row stride
  __shared__ __bf16 Qs[128][LQ];
  __shared__ __bf16 Ks[2][64][LQ];   // [key][d]  -> B frag for S = Q K^T
  __shared__ __bf16 Vts[2][64][LQ];  // [d][key]  -> B frag for O = P V
  __shared__ __bf16 Ps[8][16][LQ];   // per-wave P tile

  const int tid  = threadIdx.x;
  const int lane = tid & 31;
  const int wave = tid >> 5;
  const int nl   = lane & 15;
  const int half = lane >> 4;

  const int bh    = blockIdx.y;  // b*H + h
  const int qbase = blockIdx.x * 128;
  const size_t base = (size_t)bh * NSEQ * DH;

  auto stage_kv = [&](int buf, int kb) {
    // K tile [key][d] via async DMA.
    {
      const int r = tid >> 3, seg = (tid & 7) * 8;
#pragma unroll
      for (int p = 0; p < 2; ++p) {
        const int row = p * 32 + r;
        async_ld_b128(&Ks[buf][row][seg],
                      &Kg[base + (size_t)(kb + row) * DH + seg]);
      }
    }
    // V tile transposed [d][key] (layout change: manual stage).
    {
      const int key = tid >> 2, dseg = (tid & 3) * 16;
      const size_t src = base + (size_t)(kb + key) * DH + dseg;
      bf16x8 v0 = *(const bf16x8*)&Vg[src];
      bf16x8 v1 = *(const bf16x8*)&Vg[src + 8];
#pragma unroll
      for (int i = 0; i < 8; ++i) {
        Vts[buf][dseg + i][key]     = v0[i];
        Vts[buf][dseg + 8 + i][key] = v1[i];
      }
    }
  };

  // ---- stage Q tile (128 x 64 bf16) via async DMA + first K/V block ----
  {
    const int r = tid >> 3, seg = (tid & 7) * 8;
#pragma unroll
    for (int p = 0; p < 4; ++p) {
      const int row = p * 32 + r;
      async_ld_b128(&Qs[row][seg], &Qg[base + (size_t)(qbase + row) * DH + seg]);
    }
  }
  stage_kv(0, 0);

  float m_run[8], l_run[8];
  f32x8 o0 = zero8(), o1 = zero8(), o2 = zero8(), o3 = zero8();
#pragma unroll
  for (int r = 0; r < 8; ++r) { m_run[r] = -1e30f; l_run[r] = 0.0f; }

  int cur = 0;
  for (int kb = 0; kb < NSEQ; kb += 64) {
    wait_async0();
    __syncthreads();
    if (kb + 64 < NSEQ) stage_kv(cur ^ 1, kb + 64);  // overlap next block DMA

    // ---- S = Q K^T : wave's 16x64 score block ----
    f32x8 s[4] = {zero8(), zero8(), zero8(), zero8()};
#pragma unroll
    for (int ks = 0; ks < 2; ++ks) {
      const int k0 = ks * 32;
      const int qrow = wave * 16 + nl;
      const bf16x16 a =
          ld_frag16(&Qs[qrow][k0 + half * 8], &Qs[qrow][k0 + 16 + half * 8]);
#pragma unroll
      for (int tn = 0; tn < 4; ++tn) {
        const int kcol = tn * 16 + nl;
        const bf16x16 b = ld_frag16(&Ks[cur][kcol][k0 + half * 8],
                                    &Ks[cur][kcol][k0 + 16 + half * 8]);
        s[tn] = wmma_bf16(a, b, s[tn]);
      }
    }

    // ---- online softmax (rows live in 16-lane halves of the wave) ----
#pragma unroll
    for (int r = 0; r < 8; ++r) {
      float v0 = s[0][r] * SM_SCALE, v1 = s[1][r] * SM_SCALE;
      float v2 = s[2][r] * SM_SCALE, v3 = s[3][r] * SM_SCALE;
      float mx = fmaxf(fmaxf(v0, v1), fmaxf(v2, v3));
#pragma unroll
      for (int off = 1; off < 16; off <<= 1) mx = fmaxf(mx, __shfl_xor(mx, off, 32));
      const float mnew  = fmaxf(m_run[r], mx);
      const float alpha = __expf(m_run[r] - mnew);
      const float p0 = __expf(v0 - mnew), p1 = __expf(v1 - mnew);
      const float p2 = __expf(v2 - mnew), p3 = __expf(v3 - mnew);
      float rs = p0 + p1 + p2 + p3;
#pragma unroll
      for (int off = 1; off < 16; off <<= 1) rs += __shfl_xor(rs, off, 32);
      l_run[r] = l_run[r] * alpha + rs;
      m_run[r] = mnew;
      o0[r] *= alpha; o1[r] *= alpha; o2[r] *= alpha; o3[r] *= alpha;
      const int prow = r + half * 8;  // C-layout row
      Ps[wave][prow][ 0 + nl] = (__bf16)p0;
      Ps[wave][prow][16 + nl] = (__bf16)p1;
      Ps[wave][prow][32 + nl] = (__bf16)p2;
      Ps[wave][prow][48 + nl] = (__bf16)p3;
    }
    // wave-local LDS RAW fence (P writes -> A-frag reads)
    asm volatile("s_wait_dscnt 0" ::: "memory");

    // ---- O += P V ----
#pragma unroll
    for (int ks = 0; ks < 2; ++ks) {
      const int k0 = ks * 32;
      const bf16x16 a = ld_frag16(&Ps[wave][nl][k0 + half * 8],
                                  &Ps[wave][nl][k0 + 16 + half * 8]);
#pragma unroll
      for (int tn = 0; tn < 4; ++tn) {
        const int dcol = tn * 16 + nl;
        const bf16x16 b = ld_frag16(&Vts[cur][dcol][k0 + half * 8],
                                    &Vts[cur][dcol][k0 + 16 + half * 8]);
        f32x8* op = (tn == 0) ? &o0 : (tn == 1) ? &o1 : (tn == 2) ? &o2 : &o3;
        *op = wmma_bf16(a, b, *op);
      }
    }
    cur ^= 1;
  }

  // ---- normalize + store to [B, N, C] bf16 ----
  const int b = bh >> 4, h = bh & (HH - 1);
#pragma unroll
  for (int r = 0; r < 8; ++r) {
    const float inv = 1.0f / l_run[r];
    const int token = qbase + wave * 16 + r + half * 8;
    const size_t rowoff = ((size_t)b * NSEQ + token) * CDIM + h * DH;
    Og[rowoff +  0 + nl] = (__bf16)(o0[r] * inv);
    Og[rowoff + 16 + nl] = (__bf16)(o1[r] * inv);
    Og[rowoff + 32 + nl] = (__bf16)(o2[r] * inv);
    Og[rowoff + 48 + nl] = (__bf16)(o3[r] * inv);
  }
}

// ---------------------------------------------------------------------------
extern "C" void kernel_launch(void* const* d_in, const int* in_sizes, int n_in,
                              void* d_out, int out_size, void* d_ws, size_t ws_size,
                              hipStream_t stream) {
  const float* x  = (const float*)d_in[0];
  const float* Wq = (const float*)d_in[1];
  const float* bq = (const float*)d_in[2];
  const float* Wk = (const float*)d_in[3];
  const float* bk = (const float*)d_in[4];
  const float* Wv = (const float*)d_in[5];
  const float* bv = (const float*)d_in[6];
  const float* Wp = (const float*)d_in[7];
  const float* bp = (const float*)d_in[8];

  const size_t mtx = (size_t)BB * NSEQ * CDIM;  // 8M elems
  const size_t wsz = (size_t)CDIM * CDIM;       // 1M elems
  __bf16* xbf = (__bf16*)d_ws;   // [B*N, C] bf16
  __bf16* wqb = xbf + mtx;
  __bf16* wkb = wqb + wsz;
  __bf16* wvb = wkb + wsz;
  __bf16* wpb = wvb + wsz;
  __bf16* qws = wpb + wsz;       // [B,H,N,D]
  __bf16* kws = qws + mtx;
  __bf16* vws = kws + mtx;
  __bf16* aws = vws + mtx;       // attn out [B,N,C]

  // one-time bf16 conversion of activations + weights
  {
    const int nx4 = (int)(mtx / 4), nw4 = (int)(wsz / 4);
    cvt_f32_to_bf16<<<(nx4 + 255) / 256, 256, 0, stream>>>(x, xbf, nx4);
    cvt_f32_to_bf16<<<(nw4 + 255) / 256, 256, 0, stream>>>(Wq, wqb, nw4);
    cvt_f32_to_bf16<<<(nw4 + 255) / 256, 256, 0, stream>>>(Wk, wkb, nw4);
    cvt_f32_to_bf16<<<(nw4 + 255) / 256, 256, 0, stream>>>(Wv, wvb, nw4);
    cvt_f32_to_bf16<<<(nw4 + 255) / 256, 256, 0, stream>>>(Wp, wpb, nw4);
  }

  const int M = BB * NSEQ;
  dim3 gGemm(M / 128, CDIM / 128);
  gemm_bf16_wmma<0><<<gGemm, 256, 0, stream>>>(xbf, wqb, bq, qws, M, CDIM, CDIM);
  gemm_bf16_wmma<0><<<gGemm, 256, 0, stream>>>(xbf, wkb, bk, kws, M, CDIM, CDIM);
  gemm_bf16_wmma<0><<<gGemm, 256, 0, stream>>>(xbf, wvb, bv, vws, M, CDIM, CDIM);

  dim3 gAtt(NSEQ / 128, BB * HH);
  attn_fwd<<<gAtt, 256, 0, stream>>>(qws, kws, vws, aws);

  gemm_bf16_wmma<1><<<gGemm, 256, 0, stream>>>(aws, wpb, bp, d_out, M, CDIM, CDIM);
}
